// SelfAttention_12395275616846
// MI455X (gfx1250) — matrix operand order repaired and checked by reading, not compiled, not run on hardware
//
#include <hip/hip_runtime.h>

#define EMBED 1024
#define BATCH 4
#define SEQ   2048
#define MROWS (BATCH * SEQ) /* 8192 */

typedef __attribute__((ext_vector_type(16))) __bf16         bf16x16;
typedef __attribute__((ext_vector_type(8)))  float          v8f;
typedef __attribute__((ext_vector_type(16))) unsigned short u16x16;
typedef __attribute__((ext_vector_type(8)))  unsigned short u16x8;
typedef __attribute__((ext_vector_type(4)))  unsigned int   u32x4;
typedef __attribute__((ext_vector_type(8)))  int            i32x8;
typedef __attribute__((ext_vector_type(4)))  int            i32x4;

#if defined(__has_builtin)
#  if __has_builtin(__builtin_amdgcn_tensor_load_to_lds) && \
      __has_builtin(__builtin_amdgcn_s_wait_tensorcnt)
#    define HAVE_TDM 1
#  else
#    define HAVE_TDM 0
#  endif
#else
#  define HAVE_TDM 0
#endif

// float -> bf16, round-to-nearest-even
__device__ __forceinline__ unsigned short f2bf(float f) {
    unsigned u = __float_as_uint(f);
    u += 0x7fffu + ((u >> 16) & 1u);
    return (unsigned short)(u >> 16);
}
__device__ __forceinline__ unsigned pack2bf(float a, float b) {
    return (unsigned)f2bf(a) | ((unsigned)f2bf(b) << 16);
}

// Assemble a 16-element bf16 WMMA fragment from two 8-element (16B) chunks.
__device__ __forceinline__ bf16x16 pack_frag(u16x8 lo, u16x8 hi) {
    u16x16 u;
#pragma unroll
    for (int i = 0; i < 8; ++i) { u[i] = lo[i]; u[i + 8] = hi[i]; }
    return __builtin_bit_cast(bf16x16, u);
}

#if HAVE_TDM
// Issue a TDM 2-D bf16 tile load: global (row-major, elem stride_d0 per row)
// -> contiguous LDS at byte offset lds_off.  Dims/tiles in elements.
// 6-arg toolchain variant: (g0, g1, g2, g3, g2b, cpol); groups 2/3 zero (2-D).
__device__ __forceinline__ void tdm_load_2d(unsigned lds_off, const void* gaddr,
                                            unsigned d0, unsigned d1,
                                            unsigned t0, unsigned t1,
                                            unsigned stride_d0) {
    unsigned long long ga = (unsigned long long)(uintptr_t)gaddr;
    u32x4 g0;
    g0[0] = 1u;                                   // count=1, user descriptor
    g0[1] = lds_off;                              // lds_addr (bytes)
    g0[2] = (unsigned)ga;                         // global_addr[31:0]
    g0[3] = (unsigned)((ga >> 32) & 0x1ffffffu) | (2u << 30); // addr[56:32], type=2
    i32x8 g1;
    g1[0] = (int)(1u << 16);                      // data_size=1 -> 2 bytes
    g1[1] = (int)(d0 << 16);                      // tensor_dim0 [79:48]
    g1[2] = (int)((d0 >> 16) | (d1 << 16));       // ... tensor_dim1 [111:80]
    g1[3] = (int)((d1 >> 16) | (t0 << 16));       // ... tile_dim0 [127:112]
    g1[4] = (int)(t1 & 0xffffu);                  // tile_dim1, tile_dim2=0
    g1[5] = (int)stride_d0;                       // tensor_dim0_stride [207:160]
    g1[6] = 0;
    g1[7] = 0;
    i32x4 z4 = {};
    i32x8 z8 = {};
    __builtin_amdgcn_tensor_load_to_lds(g0, g1, z4, z4, z8, 0);
}
#endif

// ---------------------------------------------------------------------------
// Kernel 1: fused QKV projection.  y = x @ W + b, bf16 outputs.
// ---------------------------------------------------------------------------
#define SBT 40  // padded LDS row stride (ushorts) for transposed W tile

__global__ __launch_bounds__(256) void qkv_gemm(
    const float* __restrict__ x,
    const float* __restrict__ Wq, const float* __restrict__ bq,
    const float* __restrict__ Wk, const float* __restrict__ bk,
    const float* __restrict__ Wv, const float* __restrict__ bv,
    unsigned short* __restrict__ Qb,   // [MROWS, EMBED] bf16
    unsigned short* __restrict__ Kb,   // [MROWS, EMBED] bf16
    unsigned short* __restrict__ VT)   // [BATCH, EMBED, SEQ] bf16 (transposed)
{
    __shared__ __align__(16) unsigned short sA[16 * 32];    // x tile, bf16
    __shared__ __align__(16) unsigned short sBt[128 * SBT]; // W tile, bf16, transposed

    const int mat = blockIdx.z;
    const float* W    = (mat == 0) ? Wq : (mat == 1) ? Wk : Wv;
    const float* bias = (mat == 0) ? bq : (mat == 1) ? bk : bv;

    const int mbase  = blockIdx.x * 16;
    const int nblock = blockIdx.y * 128;
    const int t    = threadIdx.x;
    const int lane = t & 31, w = t >> 5;
    const int r    = lane & 15, h = lane >> 4;

    v8f acc = {};

#pragma unroll 1
    for (int k0 = 0; k0 < EMBED; k0 += 32) {
        // Stage A: x[mbase:+16, k0:+32] -> bf16, packed b32 stores
        {
            int idx = t * 2;
            int rr = idx >> 5, ee = idx & 31;
            float2 v = *(const float2*)(x + (size_t)(mbase + rr) * EMBED + k0 + ee);
            *(unsigned*)(sA + rr * 32 + ee) = pack2bf(v.x, v.y);
        }
        // Stage B transposed: W[k0:+32, nblock:+128] -> sBt[n][k]
#pragma unroll
        for (int i = 0; i < 16; ++i) {
            int idx = t + 256 * i;
            int kk = idx >> 7, nn = idx & 127;
            sBt[nn * SBT + kk] = f2bf(W[(size_t)(k0 + kk) * EMBED + nblock + nn]);
        }
        if (k0 + 32 < EMBED)  // prefetch next W tile row chunk
            __builtin_prefetch(W + (size_t)(k0 + 32) * EMBED + nblock + (t & 127), 0, 0);
        __syncthreads();

        u16x8 a0 = *(const u16x8*)(sA + r * 32 + 8 * h);
        u16x8 a1 = *(const u16x8*)(sA + r * 32 + 16 + 8 * h);
        bf16x16 afrag = pack_frag(a0, a1);
        const unsigned short* bp = sBt + (w * 16 + r) * SBT + 8 * h;
        u16x8 b0 = *(const u16x8*)(bp);
        u16x8 b1 = *(const u16x8*)(bp + 16);
        bf16x16 bfrag = pack_frag(b0, b1);

        acc = __builtin_amdgcn_wmma_f32_16x16x32_bf16(
            false, afrag, false, bfrag, (short)0, acc, false, false);
        __syncthreads();
    }

    const int ncol = nblock + w * 16 + r;
    const float bval = bias[ncol];
#pragma unroll
    for (int i = 0; i < 8; ++i) acc[i] += bval;

    if (mat < 2) {
        unsigned short* dst = (mat == 0) ? Qb : Kb;
#pragma unroll
        for (int i = 0; i < 8; ++i) {               // C layout: M = i + 8h, N = ncol
            int m = mbase + i + 8 * h;
            dst[(size_t)m * EMBED + ncol] = f2bf(acc[i]);
        }
    } else {
        int batch = mbase / SEQ;
        int s0    = (mbase - batch * SEQ) + 8 * h;
        u16x8 pk;
#pragma unroll
        for (int i = 0; i < 8; ++i) pk[i] = f2bf(acc[i]);
        *(u16x8*)(VT + (size_t)batch * EMBED * SEQ + (size_t)ncol * SEQ + s0) = pk;
    }
}

// ---------------------------------------------------------------------------
// Kernel 2: attention, one 16-query tile per wave (one wave per block).
// LDS: sQ 32K | sK 2x32K (TDM double buffer) | sS 128K | sP 64K  = 288KB
// ---------------------------------------------------------------------------
#define SQ_OFF  0
#define SK_OFF  32768
#define SS_OFF  98304
#define SP_OFF  229376
#define SMEM_SZ 294912
#define KBUF_BYTES 32768

__global__ __launch_bounds__(32) void attn_tile(
    const unsigned short* __restrict__ Qb,
    const unsigned short* __restrict__ Kb,
    const unsigned short* __restrict__ VT,
    float* __restrict__ out)
{
    extern __shared__ char smem[];
    unsigned short* sQ = (unsigned short*)(smem + SQ_OFF);   // [16][1024]
    unsigned short* sK = (unsigned short*)(smem + SK_OFF);   // 2 x [16][1024]
    float*          sS = (float*)(smem + SS_OFF);            // [16][2048]
    unsigned short* sP = (unsigned short*)(smem + SP_OFF);   // [16][2048]

    const int lane = threadIdx.x & 31;
    const int r = lane & 15, h = lane >> 4;
    const int batch = blockIdx.y;
    const int q0    = blockIdx.x * 16;
    const size_t qrow = (size_t)batch * SEQ + q0;

#if HAVE_TDM
    const unsigned lds_base = (unsigned)(uintptr_t)(void*)smem;
    // Prologue: DMA Q tile and K chunk 0 while we set up.
    tdm_load_2d(lds_base + SQ_OFF, Qb + qrow * EMBED, EMBED, 16, EMBED, 16, EMBED);
    tdm_load_2d(lds_base + SK_OFF, Kb + (size_t)batch * SEQ * EMBED,
                EMBED, 16, EMBED, 16, EMBED);
#else
#pragma unroll 1
    for (int i = 0; i < 64; ++i) {
        int idx = lane + 32 * i;
        int row = idx >> 7;
        int c   = (idx & 127) * 8;
        *(u16x8*)(sQ + row * EMBED + c) =
            *(const u16x8*)(Qb + (qrow + row) * EMBED + c);
    }
    __syncthreads();
#endif

    const float scale = 0.03125f; // 1/sqrt(1024)

    // ---- Phase 1: scores S = scale * Q @ K^T (K chunks DMA double-buffered) ----
#pragma unroll 1
    for (int kc = 0; kc < SEQ / 16; ++kc) {
#if HAVE_TDM
        if (kc + 1 < SEQ / 16) {
            tdm_load_2d(lds_base + SK_OFF + ((kc + 1) & 1) * KBUF_BYTES,
                        Kb + ((size_t)batch * SEQ + (size_t)(kc + 1) * 16) * EMBED,
                        EMBED, 16, EMBED, 16, EMBED);
            __builtin_amdgcn_s_wait_tensorcnt(1);  // current chunk (and Q) landed
        } else {
            __builtin_amdgcn_s_wait_tensorcnt(0);
        }
        asm volatile("" ::: "memory");
        const unsigned short* kbuf = sK + (kc & 1) * (16 * EMBED);
#else
#pragma unroll 1
        for (int i = 0; i < 64; ++i) {             // stage chunk kc manually
            int idx = lane + 32 * i;
            int row = idx >> 7;
            int c   = (idx & 127) * 8;
            *(u16x8*)(sK + row * EMBED + c) =
                *(const u16x8*)(Kb + ((size_t)batch * SEQ + kc * 16 + row) * EMBED + c);
        }
        __syncthreads();
        const unsigned short* kbuf = sK;
#endif
        const unsigned short* kbase = kbuf + r * EMBED;  // key row = column n
        v8f acc0 = {}, acc1 = {};
#pragma unroll 1
        for (int e0 = 0; e0 < EMBED; e0 += 64) {   // two independent WMMA chains
            u16x8 a0 = *(const u16x8*)(sQ + r * EMBED + e0 + 8 * h);
            u16x8 a1 = *(const u16x8*)(sQ + r * EMBED + e0 + 16 + 8 * h);
            u16x8 b0 = *(const u16x8*)(kbase + e0 + 8 * h);
            u16x8 b1 = *(const u16x8*)(kbase + e0 + 16 + 8 * h);
            acc0 = __builtin_amdgcn_wmma_f32_16x16x32_bf16(
                false, pack_frag(a0, a1), false, pack_frag(b0, b1),
                (short)0, acc0, false, false);
            u16x8 c0 = *(const u16x8*)(sQ + r * EMBED + e0 + 32 + 8 * h);
            u16x8 c1 = *(const u16x8*)(sQ + r * EMBED + e0 + 48 + 8 * h);
            u16x8 d0 = *(const u16x8*)(kbase + e0 + 32 + 8 * h);
            u16x8 d1 = *(const u16x8*)(kbase + e0 + 48 + 8 * h);
            acc1 = __builtin_amdgcn_wmma_f32_16x16x32_bf16(
                false, pack_frag(c0, c1), false, pack_frag(d0, d1),
                (short)0, acc1, false, false);
        }
        v8f acc = acc0 + acc1;
#pragma unroll
        for (int i = 0; i < 8; ++i)  // C layout: M = i+8h (query), N = r (key)
            sS[(i + 8 * h) * SEQ + kc * 16 + r] = acc[i] * scale;
#if !HAVE_TDM
        __syncthreads();
#endif
    }
    __syncthreads();

    // ---- Phase 2: row softmax (lane = half-row; combine halves via shfl) ----
    {
        float* rowp = sS + r * SEQ + h * 1024;
        float m = -1e30f;
#pragma unroll 1
        for (int j = 0; j < 1024; ++j) m = fmaxf(m, rowp[j]);
        m = fmaxf(m, __shfl_xor(m, 16));
        float l = 0.f;
#pragma unroll 1
        for (int j = 0; j < 1024; ++j) {
            float p = __expf(rowp[j] - m);
            rowp[j] = p;
            l += p;
        }
        l += __shfl_xor(l, 16);
        float inv = 1.f / l;
        unsigned short* prow = sP + r * SEQ + h * 1024;
#pragma unroll 1
        for (int j = 0; j < 1024; ++j) prow[j] = f2bf(rowp[j] * inv);
    }
    __syncthreads();

    // ---- Phase 3: O = P @ V (V^T rows give contiguous B-fragments) ----
    const unsigned short* vb = VT + (size_t)batch * EMBED * SEQ;
#pragma unroll 1
    for (int et = 0; et < EMBED / 16; ++et) {
        v8f acc0 = {}, acc1 = {};
        const unsigned short* vcol = vb + (size_t)(et * 16 + r) * SEQ; // e column = n
#pragma unroll 1
        for (int kk = 0; kk < SEQ; kk += 64) {
            u16x8 a0 = *(const u16x8*)(sP + r * SEQ + kk + 8 * h);
            u16x8 a1 = *(const u16x8*)(sP + r * SEQ + kk + 16 + 8 * h);
            u16x8 b0 = *(const u16x8*)(vcol + kk + 8 * h);
            u16x8 b1 = *(const u16x8*)(vcol + kk + 16 + 8 * h);
            acc0 = __builtin_amdgcn_wmma_f32_16x16x32_bf16(
                false, pack_frag(a0, a1), false, pack_frag(b0, b1),
                (short)0, acc0, false, false);
            u16x8 c0 = *(const u16x8*)(sP + r * SEQ + kk + 32 + 8 * h);
            u16x8 c1 = *(const u16x8*)(sP + r * SEQ + kk + 48 + 8 * h);
            u16x8 d0 = *(const u16x8*)(vcol + kk + 32 + 8 * h);
            u16x8 d1 = *(const u16x8*)(vcol + kk + 48 + 8 * h);
            acc1 = __builtin_amdgcn_wmma_f32_16x16x32_bf16(
                false, pack_frag(c0, c1), false, pack_frag(d0, d1),
                (short)0, acc1, false, false);
        }
        v8f acc = acc0 + acc1;
#pragma unroll
        for (int i = 0; i < 8; ++i)
            out[(qrow + i + 8 * h) * EMBED + et * 16 + r] = acc[i];
    }
}

// ---------------------------------------------------------------------------
extern "C" void kernel_launch(void* const* d_in, const int* in_sizes, int n_in,
                              void* d_out, int out_size, void* d_ws, size_t ws_size,
                              hipStream_t stream) {
    const float* x  = (const float*)d_in[0];
    const float* Wq = (const float*)d_in[1];
    const float* bq = (const float*)d_in[2];
    const float* Wk = (const float*)d_in[3];
    const float* bk = (const float*)d_in[4];
    const float* Wv = (const float*)d_in[5];
    const float* bv = (const float*)d_in[6];
    float* out = (float*)d_out;

    unsigned short* Qb = (unsigned short*)d_ws;
    unsigned short* Kb = Qb + (size_t)MROWS * EMBED;
    unsigned short* VT = Kb + (size_t)MROWS * EMBED;

    dim3 g1(MROWS / 16, EMBED / 128, 3);
    qkv_gemm<<<g1, 256, 0, stream>>>(x, Wq, bq, Wk, bk, Wv, bv, Qb, Kb, VT);

    dim3 g2(SEQ / 16, BATCH);
    attn_tile<<<g2, 32, SMEM_SZ, stream>>>(Qb, Kb, VT, out);
}
